// DU_LM_SB_38860864094329
// MI455X (gfx1250) — matrix operand order repaired and checked by reading, not compiled, not run on hardware
//
#include <hip/hip_runtime.h>
#include <hip/hip_bf16.h>
#include <math.h>

// ---------------------------------------------------------------------------
// DU-LM-SB MIMO detector for MI455X (gfx1250), wave32 + WMMA bf16.
//
// Sizes (fixed by harness): Nr=Nt=128 -> N=256, nbps=6 -> rb=3, Ns=768,
// B=16384, T=100, M=64, qam_var=42, powers s={4,2,1}.
//
// Phase 1 (tiny): H_t, A=H_t H_t^T + lam I, Gauss-Jordan inverse in LDS,
//   G = H_t^T A^{-1} H_t / lam, h, c0, and J packed to bf16 hi/lo fragments.
// Phase 2 (dominant): 512 workgroups, each owns 32 batch columns; x, y and
//   the bf16 packed matmul operand live in LDS for all T iterations.
//   J@x via v_wmma_f32_16x16x32_bf16 with split-bf16 (3 products).
// ---------------------------------------------------------------------------

typedef __bf16 v16bf __attribute__((ext_vector_type(16)));
typedef float  v8f   __attribute__((ext_vector_type(8)));

#define NS   768
#define NN   256
#define BTOT 16384
#define BT   32          // columns per workgroup
#define NWG  (BTOT / BT) // 512
#define KT_N 24          // 768 / 32 k-steps
#define MT_N 48          // 768 / 16 m-tiles
// packed J fragments: [(kt*48+mt)*32 + lane]*16 bf16 elements
#define JPACK_ELEMS (MT_N * KT_N * 32 * 16)  // 589824

// ---- workspace float-offsets -----------------------------------------------
#define OFF_HT    0        // 65536
#define OFF_AINV  65536    // 65536
#define OFF_V     131072   // 65536
#define OFF_G     196608   // 65536
#define OFF_YT    262144   // 256
#define OFF_W     262400   // 256
#define OFF_H     262656   // 768
#define OFF_SUMS  263424   // 2
#define OFF_ETAC0 263426   // 1
#define OFF_JHI_BYTES 1056768u            // = 264192 floats * 4
#define OFF_JLO_BYTES (1056768u + 1179648u)

__device__ __forceinline__ float sigm(float v) { return 1.0f / (1.0f + __expf(-v)); }
__device__ __forceinline__ float swish(float v) { return v * sigm(v); }

// ---------------------------------------------------------------------------
// K1: build H_t = [[Hr,-Hi],[Hi,Hr]] and y_t = [yr;yi]
__global__ void k_build_ht(const float* __restrict__ Hr, const float* __restrict__ Hi,
                           const float* __restrict__ yr, const float* __restrict__ yi,
                           float* __restrict__ Ht, float* __restrict__ yt) {
  int idx = blockIdx.x * 256 + threadIdx.x;
  if (idx >= NN * NN) return;
  int i = idx >> 8, j = idx & 255;
  float v;
  if (i < 128) v = (j < 128) ? Hr[i * 128 + j] : -Hi[i * 128 + (j - 128)];
  else         v = (j < 128) ? Hi[(i - 128) * 128 + j] : Hr[(i - 128) * 128 + (j - 128)];
  Ht[idx] = v;
  if (j == 0) yt[i] = (i < 128) ? yr[i] : yi[i - 128];
}

// ---------------------------------------------------------------------------
// K2: A = Ht*Ht^T + lam*I built in LDS (row stride 257 -> conflict free),
// in-place Gauss-Jordan inversion (SPD + lam*I: no pivoting), single WG.
__global__ void k_gj_inverse(const float* __restrict__ Ht, const float* __restrict__ lam_p,
                             float* __restrict__ Ainv) {
  extern __shared__ __align__(32) char smem_gj[];
  float* As   = (float*)smem_gj;          // 256*257
  float* rowk = As + NN * 257;            // 256
  int tid = threadIdx.x;                  // 256 threads
  float lam = *lam_p;

  for (int idx = tid; idx < NN * NN; idx += 256) {
    int i = idx >> 8, j = idx & 255;
    float s = (i == j) ? lam : 0.0f;
    for (int k = 0; k < NN; ++k) s += Ht[i * NN + k] * Ht[j * NN + k];
    As[i * 257 + j] = s;
  }
  __syncthreads();

  for (int k = 0; k < NN; ++k) {
    float ip = 1.0f / As[k * 257 + k];
    rowk[tid] = As[k * 257 + tid] * ip;
    __syncthreads();
    int i = tid;
    if (i == k) {
      for (int j = 0; j < NN; ++j) As[k * 257 + j] = (j == k) ? ip : rowk[j];
    } else {
      float f = As[i * 257 + k];
      for (int j = 0; j < NN; ++j)
        As[i * 257 + j] = (j == k) ? (-f * ip) : (As[i * 257 + j] - f * rowk[j]);
    }
    __syncthreads();
  }

  for (int idx = tid; idx < NN * NN; idx += 256) {
    int i = idx >> 8, j = idx & 255;
    Ainv[idx] = As[i * 257 + j];
  }
}

// ---------------------------------------------------------------------------
// K3: V = Ainv @ Ht ; w = Ainv @ (yt/sqrt(42)) / lam   (z-term collapses)
__global__ void k_V_w(const float* __restrict__ Ainv, const float* __restrict__ Ht,
                      const float* __restrict__ yt, const float* __restrict__ lam_p,
                      float* __restrict__ V, float* __restrict__ w) {
  int idx = blockIdx.x * 256 + threadIdx.x;
  if (idx >= NN * NN) return;
  int i = idx >> 8, j = idx & 255;
  float s = 0.0f;
  for (int k = 0; k < NN; ++k) s += Ainv[i * NN + k] * Ht[k * NN + j];
  V[idx] = s;
  if (j == 0) {
    float lam = *lam_p;
    float sw_ = 0.0f;
    for (int k = 0; k < NN; ++k) sw_ += Ainv[i * NN + k] * yt[k];
    w[i] = sw_ / (lam * sqrtf(42.0f));
  }
}

__global__ void k_zero_sums(float* __restrict__ sums) {
  if (threadIdx.x < 2) sums[threadIdx.x] = 0.0f;
}

// ---------------------------------------------------------------------------
// K4: G = Ht^T @ V / lam ; accumulate ||G||^2 and sum G_ii^2
__global__ void k_G(const float* __restrict__ Ht, const float* __restrict__ V,
                    const float* __restrict__ lam_p, float* __restrict__ G,
                    float* __restrict__ sums) {
  __shared__ float r0[256], r1[256];
  int tid = threadIdx.x;
  int idx = blockIdx.x * 256 + tid;
  int i = idx >> 8, j = idx & 255;
  float lam = *lam_p;
  float s = 0.0f;
  for (int k = 0; k < NN; ++k) s += Ht[k * NN + i] * V[k * NN + j];
  s /= lam;
  G[idx] = s;
  r0[tid] = s * s;
  r1[tid] = (i == j) ? s * s : 0.0f;
  __syncthreads();
  for (int off = 128; off > 0; off >>= 1) {
    if (tid < off) { r0[tid] += r0[tid + off]; r1[tid] += r1[tid + off]; }
    __syncthreads();
  }
  if (tid == 0) { atomicAdd(&sums[0], r0[0]); atomicAdd(&sums[1], r1[0]); }
}

// ---------------------------------------------------------------------------
// K5: c0; sumJ^2 = (2/42)^2 * (21^2 * ||G||^2 - 273 * sum G_ii^2)
__global__ void k_c0(const float* __restrict__ sums, const float* __restrict__ eta_p,
                     float* __restrict__ etac0) {
  const float coef = 2.0f / 42.0f;
  float sumJ2 = coef * coef * (441.0f * sums[0] - 273.0f * sums[1]);
  float c0 = 2.0f * sqrtf((float)(NS - 1) / sumJ2);
  *etac0 = eta_p[0] * c0;
}

// K6: h[r*256+i] = 2 * s_r * (Ht^T w)[i],  s = {4,2,1}
__global__ void k_h(const float* __restrict__ Ht, const float* __restrict__ w,
                    float* __restrict__ hvec) {
  int i = threadIdx.x;  // 256 threads, 1 block
  if (i >= NN) return;
  float s = 0.0f;
  for (int k = 0; k < NN; ++k) s += Ht[k * NN + i] * w[k];
  hvec[i]           = 8.0f * s;   // 2*4
  hvec[NN + i]      = 4.0f * s;   // 2*2
  hvec[2 * NN + i]  = 2.0f * s;   // 2*1
}

// ---------------------------------------------------------------------------
// K7: pack J = -(2/42)*s_r*s_q*G (diag zeroed) into bf16 hi/lo A-fragments.
// A 16x32 bf16 lane layout: lane<16 -> K={kb..kb+15}, kb=0;
//                           lane>=16 -> kb=16.  One lane = 16 contiguous bf16.
__global__ void k_packJ(const float* __restrict__ G, __bf16* __restrict__ Jhi,
                        __bf16* __restrict__ Jlo) {
  int idx = blockIdx.x * 256 + threadIdx.x;  // (kt*48+mt)*32 + lane
  if (idx >= MT_N * KT_N * 32) return;
  int lane = idx & 31;
  int frag = idx >> 5;
  int mt = frag % MT_N;
  int kt = frag / MT_N;
  int M  = mt * 16 + (lane & 15);
  int kb = kt * 32 + ((lane < 16) ? 0 : 16);
  int r = M >> 8, i = M & 255;
  float sr = (float)(1 << (2 - r));
  const float scale = -2.0f / 42.0f;
  for (int e = 0; e < 16; ++e) {
    int K = kb + e;
    int q = K >> 8, j = K & 255;
    float sq = (float)(1 << (2 - q));
    float val = (M == K) ? 0.0f : scale * sr * sq * G[i * NN + j];
    __bf16 hb = (__bf16)val;
    Jhi[idx * 16 + e] = hb;
    Jlo[idx * 16 + e] = (__bf16)(val - (float)hb);
  }
}

// ---------------------------------------------------------------------------
// K8: main SB loop. 512 WGs x 256 threads (8 waves). Everything in LDS.
__global__ __launch_bounds__(256, 1)
void k_sb_main(const __bf16* __restrict__ Jhi, const __bf16* __restrict__ Jlo,
               const float* __restrict__ hvec, const float* __restrict__ etac0p,
               const float* __restrict__ delta, const float* __restrict__ x0,
               const float* __restrict__ y0, float* __restrict__ out, int T) {
  extern __shared__ __align__(32) char smem[];
  __bf16* xph = (__bf16*)smem;                       // 24*2*32*16 = 24576 bf16
  __bf16* xpl = xph + KT_N * 2 * 32 * 16;            // 24576 bf16
  float*  xs  = (float*)(xpl + KT_N * 2 * 32 * 16);  // 768*33
  float*  ys  = xs + NS * 33;                        // 768*33
  float*  hs  = ys + NS * 33;                        // 768

  const int cb    = blockIdx.x;        // column block (32 cols)
  const int tid   = threadIdx.x;
  const int wave  = tid >> 5;
  const int lane  = tid & 31;
  const int lhalf = lane >> 4;
  const int l16   = lane & 15;
  const float etac0 = *etac0p;

  // ---- init: x = 0.02*(x0-0.5), y = 0.02*(y0-0.5); build bf16 packs -------
  for (int idx = tid; idx < NS; idx += 256) hs[idx] = hvec[idx];
  for (int idx = tid; idx < NS * BT; idx += 256) {
    int i = idx >> 5, c = idx & 31;
    size_t g = (size_t)i * BTOT + (size_t)cb * BT + c;
    float xv = 0.02f * (x0[g] - 0.5f);
    float yv = 0.02f * (y0[g] - 0.5f);
    xs[i * 33 + c] = xv;
    ys[i * 33 + c] = yv;
    int kt = i >> 5, rem = i & 31, e = rem & 15, hf = rem >> 4;
    int nt = c >> 4;
    int L  = hf * 16 + (c & 15);
    int p  = ((kt * 2 + nt) * 32 + L) * 16 + e;
    __bf16 hb = (__bf16)xv;
    xph[p] = hb;
    xpl[p] = (__bf16)(xv - (float)hb);
  }
  __syncthreads();

  // ---- T iterations --------------------------------------------------------
  for (int t = 0; t < T; ++t) {
    float dk = delta[t];
    float ak = (float)t / (float)(T - 1);

    v8f acc[6][2];
#pragma unroll
    for (int mi = 0; mi < 6; ++mi)
#pragma unroll
      for (int nt = 0; nt < 2; ++nt)
#pragma unroll
        for (int r = 0; r < 8; ++r) acc[mi][nt][r] = 0.0f;

    for (int kt = 0; kt < KT_N; ++kt) {
      v16bf bh[2], bl[2];
#pragma unroll
      for (int nt = 0; nt < 2; ++nt) {
        int bidx = ((kt * 2 + nt) * 32 + lane) * 16;
        bh[nt] = *(const v16bf*)(xph + bidx);
        bl[nt] = *(const v16bf*)(xpl + bidx);
      }
#pragma unroll
      for (int mi = 0; mi < 6; ++mi) {
        int mt = wave * 6 + mi;
        int aoff = ((kt * MT_N + mt) * 32 + lane) * 16;
        v16bf ah = *(const v16bf*)(Jhi + aoff);
        v16bf al = *(const v16bf*)(Jlo + aoff);
#pragma unroll
        for (int nt = 0; nt < 2; ++nt) {
          acc[mi][nt] = __builtin_amdgcn_wmma_f32_16x16x32_bf16(
              false, ah, false, bh[nt], (short)0, acc[mi][nt], false, false);
          acc[mi][nt] = __builtin_amdgcn_wmma_f32_16x16x32_bf16(
              false, al, false, bh[nt], (short)0, acc[mi][nt], false, false);
          acc[mi][nt] = __builtin_amdgcn_wmma_f32_16x16x32_bf16(
              false, ah, false, bl[nt], (short)0, acc[mi][nt], false, false);
        }
      }
    }
    __syncthreads();  // all waves done reading xph/xpl

    // epilogue: D layout: VGPR r -> row mt*16 + r + 8*(lane>=16), col = nt*16 + lane%16
#pragma unroll
    for (int mi = 0; mi < 6; ++mi) {
      int mt = wave * 6 + mi;
#pragma unroll
      for (int nt = 0; nt < 2; ++nt) {
        int c = nt * 16 + l16;
#pragma unroll
        for (int r = 0; r < 8; ++r) {
          int i = mt * 16 + r + 8 * lhalf;
          float v  = acc[mi][nt][r];
          float xo = xs[i * 33 + c];
          float yo = ys[i * 33 + c];
          float yv = yo + dk * (-(1.0f - ak) * xo + etac0 * (v + hs[i]));
          float xn = xo + dk * yv;
          xn = swish(xn + 1.0f) - swish(xn - 1.0f) - 1.0f;
          yv *= (1.0f - sigm(100.0f * (fabsf(xn) - 1.01f)));
          xs[i * 33 + c] = xn;
          ys[i * 33 + c] = yv;
          int kt2 = i >> 5, rem = i & 31, e = rem & 15, hf = rem >> 4;
          int L = hf * 16 + l16;
          int p = ((kt2 * 2 + nt) * 32 + L) * 16 + e;
          __bf16 hb = (__bf16)xn;
          xph[p] = hb;
          xpl[p] = (__bf16)(xn - (float)hb);
        }
      }
    }
    __syncthreads();  // packs ready for next iteration
  }

  // ---- output: out[b][i] = x[i][b]  (B, Ns) -------------------------------
  for (int idx = tid; idx < NS * BT; idx += 256) {
    int i = idx >> 5, c = idx & 31;
    out[(size_t)(cb * BT + c) * NS + i] = xs[i * 33 + c];
  }
}

// ---------------------------------------------------------------------------
extern "C" void kernel_launch(void* const* d_in, const int* in_sizes, int n_in,
                              void* d_out, int out_size, void* d_ws, size_t ws_size,
                              hipStream_t stream) {
  const float* Hr    = (const float*)d_in[0];
  const float* Hi    = (const float*)d_in[1];
  const float* yr    = (const float*)d_in[2];
  const float* yi    = (const float*)d_in[3];
  const float* delta = (const float*)d_in[4];
  const float* eta   = (const float*)d_in[5];
  const float* lam   = (const float*)d_in[6];
  const float* x0    = (const float*)d_in[7];
  const float* y0    = (const float*)d_in[8];
  const int T = in_sizes[4];

  float* wsf   = (float*)d_ws;
  float* Ht    = wsf + OFF_HT;
  float* Ainv  = wsf + OFF_AINV;
  float* V     = wsf + OFF_V;
  float* G     = wsf + OFF_G;
  float* yt    = wsf + OFF_YT;
  float* w     = wsf + OFF_W;
  float* hvec  = wsf + OFF_H;
  float* sums  = wsf + OFF_SUMS;
  float* etac0 = wsf + OFF_ETAC0;
  __bf16* Jhi  = (__bf16*)((char*)d_ws + OFF_JHI_BYTES);
  __bf16* Jlo  = (__bf16*)((char*)d_ws + OFF_JLO_BYTES);

  const int lds_gj   = NN * 257 * 4 + NN * 4;                          // 264192
  const int lds_main = 2 * (KT_N * 2 * 32 * 16) * 2 + 2 * (NS * 33) * 4 + NS * 4;  // 304128

  (void)hipFuncSetAttribute((const void*)k_gj_inverse,
                            hipFuncAttributeMaxDynamicSharedMemorySize, lds_gj);
  (void)hipFuncSetAttribute((const void*)k_sb_main,
                            hipFuncAttributeMaxDynamicSharedMemorySize, lds_main);

  k_build_ht<<<256, 256, 0, stream>>>(Hr, Hi, yr, yi, Ht, yt);
  k_gj_inverse<<<1, 256, lds_gj, stream>>>(Ht, lam, Ainv);
  k_V_w<<<256, 256, 0, stream>>>(Ainv, Ht, yt, lam, V, w);
  k_zero_sums<<<1, 32, 0, stream>>>(sums);
  k_G<<<256, 256, 0, stream>>>(Ht, V, lam, G, sums);
  k_c0<<<1, 1, 0, stream>>>(sums, eta, etac0);
  k_h<<<1, 256, 0, stream>>>(Ht, w, hvec);
  k_packJ<<<(MT_N * KT_N * 32 + 255) / 256, 256, 0, stream>>>(G, Jhi, Jlo);
  k_sb_main<<<NWG, 256, lds_main, stream>>>(Jhi, Jlo, hvec, etac0, delta,
                                            x0, y0, (float*)d_out, T);
}
